// DySAT_44220983280300
// MI455X (gfx1250) — matrix operand order repaired and checked by reading, not compiled, not run on hardware
//
#include <hip/hip_runtime.h>
#include <hip/hip_bf16.h>

typedef __attribute__((ext_vector_type(16))) _Float16 v16h;
typedef __attribute__((ext_vector_type(8)))  _Float16 v8h;
typedef __attribute__((ext_vector_type(8)))  float    v8f;

#define NEG_SLOPE 0.2f

// ---------------------------------------------------------------------------
// Convert f32 -> f16, 4 elements per thread (n multiple of 4 by construction).
// ---------------------------------------------------------------------------
__global__ void gat_cvt_f16(const float* __restrict__ in,
                            _Float16* __restrict__ out, size_t n4) {
  const size_t tid = (size_t)blockIdx.x * blockDim.x + threadIdx.x;
  if (tid >= n4) return;
  const float4 v = *(const float4*)(in + tid * 4);
  v8h* o2 = nullptr; (void)o2;
  _Float16* o = out + tid * 4;
  o[0] = (_Float16)v.x; o[1] = (_Float16)v.y;
  o[2] = (_Float16)v.z; o[3] = (_Float16)v.w;
}

// ---------------------------------------------------------------------------
// Pack W[128,128] f32 (row-major, K major) into WMMA B-fragment order, f16.
// Fragment f = kt*8+jt (kt: K-tile of 32, jt: 16-col tile == head).
// Lane l (col N=l&15, half=l>>4) holds halfwords i in [0,16): K = 16*(i>>3) + 8*half + (i&7).
// Wp[((f*32 + lane)*16) + i] = (f16) W[(kt*32 + 16*(i>>3) + 8*half + (i&7))*128 + jt*16 + col]
// Total 4*8*32*16 = 16384 halfwords.
// ---------------------------------------------------------------------------
__global__ void gat_pack_w(const float* __restrict__ W,
                           _Float16* __restrict__ Wp) {
  const int tid = blockIdx.x * blockDim.x + threadIdx.x;   // 16384 threads
  if (tid >= 16384) return;
  const int i    = tid & 15;
  const int lane = (tid >> 4) & 31;
  const int f    = tid >> 9;
  const int kt = f >> 3, jt = f & 7;
  const int half = lane >> 4, col = lane & 15;
  const int k = kt * 32 + 16 * (i >> 3) + 8 * half + (i & 7);
  Wp[tid] = (_Float16)W[(size_t)k * 128 + jt * 16 + col];
}

// ---------------------------------------------------------------------------
// GEMM: Hout[N,128] = A[N,128](f16) @ W(pre-packed f16 fragments).
// Block = 256 threads = 8 waves; wave w owns rows [blk*128 + w*16, +16).
// Per wave: 4 K-steps x 8 column tiles = 32 v_wmma_f32_16x16x32_f16.
// A 16x32 f16 fragment (ISA 7.12.2): lane l (row M=l&15, half=l>>4) holds
//   halfwords i: K = 16*(i>>3) + 8*half + (i&7)  -> two aligned 16B runs.
// C/D 16x16 f32: VGPR r, lane l -> (M = r + 8*half, N = l&15).
// ---------------------------------------------------------------------------
__global__ void gat_gemm_wmma(const _Float16* __restrict__ Ah,
                              const _Float16* __restrict__ Wp,
                              float* __restrict__ Hout, int Nn) {
  const int lane = threadIdx.x & 31;
  const int wave = threadIdx.x >> 5;
  const int m0 = blockIdx.x * 128 + wave * 16;
  if (m0 >= Nn) return;                       // wave-uniform: EXEC stays full
  const int half = lane >> 4;
  const int col  = lane & 15;
  int mrow = m0 + col;
  if (mrow >= Nn) mrow = Nn - 1;              // clamp loads; stores guarded

  v8f acc[8];
#pragma unroll
  for (int jt = 0; jt < 8; ++jt) acc[jt] = (v8f){0.f,0.f,0.f,0.f,0.f,0.f,0.f,0.f};

#pragma unroll
  for (int kt = 0; kt < 4; ++kt) {
    // ---- A fragment: two contiguous 16B (8 x f16) loads ----
    const v8h* ab = (const v8h*)(Ah + (size_t)mrow * 128 + kt * 32 + half * 8);
    const v8h alo = ab[0];   // K = 8*half + (0..7)
    const v8h ahi = ab[2];   // K = 16 + 8*half + (0..7)
    const v16h afrag = __builtin_shufflevector(alo, ahi,
        0, 1, 2, 3, 4, 5, 6, 7, 8, 9, 10, 11, 12, 13, 14, 15);
#pragma unroll
    for (int jt = 0; jt < 8; ++jt) {
      // ---- B fragment: one contiguous 32B/lane read, wave-coalesced ----
      const v16h bfrag =
          *(const v16h*)(Wp + (((size_t)(kt * 8 + jt) * 32 + lane) << 4));
      acc[jt] = __builtin_amdgcn_wmma_f32_16x16x32_f16(
          false, afrag, false, bfrag, (short)0, acc[jt], false, false);
    }
  }

#pragma unroll
  for (int jt = 0; jt < 8; ++jt) {
#pragma unroll
    for (int r = 0; r < 8; ++r) {
      const int row = m0 + r + 8 * half;
      if (row < Nn) Hout[(size_t)row * 128 + jt * 16 + col] = acc[jt][r];
    }
  }
}

// a_src[n,h] = sum_c h[n,h,c]*att_src[h,c]; same for a_dst. 1 thread / (n,head).
__global__ void gat_coeffs(const float* __restrict__ Hh,
                           const float* __restrict__ att_s,
                           const float* __restrict__ att_d,
                           float* __restrict__ asrc, float* __restrict__ adst,
                           int Nn) {
  const int tid = blockIdx.x * blockDim.x + threadIdx.x;
  if (tid >= Nn * 8) return;
  const int head = tid & 7;
  const int n = tid >> 3;
  const float* hp = Hh + (size_t)n * 128 + head * 16;
  const float* sp = att_s + head * 16;
  const float* dp = att_d + head * 16;
  float vs = 0.f, vd = 0.f;
#pragma unroll
  for (int c = 0; c < 16; ++c) { const float hv = hp[c]; vs += hv * sp[c]; vd += hv * dp[c]; }
  asrc[tid] = vs;
  adst[tid] = vd;
}

__global__ void gat_init(float* __restrict__ mbuf, float* __restrict__ den,
                         float* __restrict__ acc, int NH, int N128) {
  const int tid = blockIdx.x * blockDim.x + threadIdx.x;
  if (tid < NH) { mbuf[tid] = __int_as_float(0xff800000); den[tid] = 0.f; }
  if (tid < N128) acc[tid] = 0.f;
}

__device__ inline void atomicMaxF(float* addr, float val) {
  if (val >= 0.f) atomicMax((int*)addr, __float_as_int(val));
  else            atomicMin((unsigned int*)addr, __float_as_uint(val));
}

// e = leaky_relu(a_src[src]+a_dst[dst]); store; atomic max into m[dst,head].
__global__ void gat_edge_max(const int* __restrict__ ei, int E, int Nn,
                             const float* __restrict__ asrc,
                             const float* __restrict__ adst,
                             float* __restrict__ ebuf, float* __restrict__ mbuf) {
  const long long tid = (long long)blockIdx.x * blockDim.x + threadIdx.x;
  const long long tot = (long long)(E + Nn) * 8;
  if (tid >= tot) return;
  const int head = (int)(tid & 7);
  const long long e = tid >> 3;
  int s, d;
  if (e < E) { s = ei[e]; d = ei[(size_t)E + e]; }
  else       { s = d = (int)(e - E); }             // self loop
  float v = asrc[(size_t)s * 8 + head] + adst[(size_t)d * 8 + head];
  v = (v > 0.f) ? v : NEG_SLOPE * v;
  ebuf[tid] = v;
  atomicMaxF(&mbuf[(size_t)d * 8 + head], v);
}

// ex = exp(e - m[dst]); store in place; atomic add into denom[dst,head].
__global__ void gat_edge_exp(const int* __restrict__ ei, int E, int Nn,
                             const float* __restrict__ mbuf,
                             float* __restrict__ ebuf, float* __restrict__ den) {
  const long long tid = (long long)blockIdx.x * blockDim.x + threadIdx.x;
  const long long tot = (long long)(E + Nn) * 8;
  if (tid >= tot) return;
  const int head = (int)(tid & 7);
  const long long e = tid >> 3;
  const int d = (e < E) ? ei[(size_t)E + e] : (int)(e - E);
  const float ex = __expf(ebuf[tid] - mbuf[(size_t)d * 8 + head]);
  ebuf[tid] = ex;
  atomicAdd(&den[(size_t)d * 8 + head], ex);
}

// acc[dst,:] += (ex/denom[dst]) * h[src,:]. 1 thread / (edge, float4 chunk).
__global__ void gat_scatter(const int* __restrict__ ei, int E, int Nn,
                            const float* __restrict__ ebuf,
                            const float* __restrict__ den,
                            const float* __restrict__ Hh,
                            float* __restrict__ acc) {
  const long long tid = (long long)blockIdx.x * blockDim.x + threadIdx.x;
  const long long tot = (long long)(E + Nn) * 32;
  if (tid >= tot) return;
  const int q = (int)(tid & 31);                 // 32 float4 chunks of 128
  const long long e = tid >> 5;
  const int head = q >> 2;                       // 16 ch per head = 4 chunks
  int s, d;
  if (e < E) { s = ei[e]; d = ei[(size_t)E + e]; }
  else       { s = d = (int)(e - E); }
  const float alpha = ebuf[(size_t)e * 8 + head] / den[(size_t)d * 8 + head];
  const float4 hv = *(const float4*)(Hh + (size_t)s * 128 + q * 4);
  float* outp = acc + (size_t)d * 128 + q * 4;
  atomicAdd(outp + 0, alpha * hv.x);
  atomicAdd(outp + 1, alpha * hv.y);
  atomicAdd(outp + 2, alpha * hv.z);
  atomicAdd(outp + 3, alpha * hv.w);
}

// Layer-1 epilogue: out_f16 = relu(acc + bias). Feeds layer-2 WMMA directly.
__global__ void gat_finalize_f16(const float* __restrict__ acc,
                                 const float* __restrict__ bias,
                                 _Float16* __restrict__ outp, int Nn) {
  const int tid = blockIdx.x * blockDim.x + threadIdx.x;
  if (tid >= Nn * 128) return;
  float v = acc[tid] + bias[tid & 127];
  v = v > 0.f ? v : 0.f;
  outp[tid] = (_Float16)v;
}

// Layer-2 epilogue: out_f32 = acc + bias.
__global__ void gat_finalize_f32(const float* __restrict__ acc,
                                 const float* __restrict__ bias,
                                 float* __restrict__ outp, int Nn) {
  const int tid = blockIdx.x * blockDim.x + threadIdx.x;
  if (tid >= Nn * 128) return;
  outp[tid] = acc[tid] + bias[tid & 127];
}

extern "C" void kernel_launch(void* const* d_in, const int* in_sizes, int n_in,
                              void* d_out, int out_size, void* d_ws, size_t ws_size,
                              hipStream_t stream) {
  const float* x   = (const float*)d_in[0];
  const int*   ei  = (const int*)d_in[1];
  const float* W1  = (const float*)d_in[2];
  const float* as1 = (const float*)d_in[3];
  const float* ad1 = (const float*)d_in[4];
  const float* b1  = (const float*)d_in[5];
  const float* W2  = (const float*)d_in[6];
  const float* as2 = (const float*)d_in[7];
  const float* ad2 = (const float*)d_in[8];
  const float* b2  = (const float*)d_in[9];

  const int Nn = in_sizes[0] / 128;
  const int E  = in_sizes[1] / 2;

  // ---- workspace layout (byte carve, base is 256B-aligned) ----
  char* p = (char*)d_ws;
  const size_t N128 = (size_t)Nn * 128;
  const size_t NH   = (size_t)Nn * 8;
  float* hbuf = (float*)p;            p += N128 * sizeof(float);   // [N,128] h
  float* accb = (float*)p;            p += N128 * sizeof(float);   // [N,128]
  float* asrc = (float*)p;            p += NH * sizeof(float);
  float* adst = (float*)p;            p += NH * sizeof(float);
  float* mbuf = (float*)p;            p += NH * sizeof(float);
  float* den  = (float*)p;            p += NH * sizeof(float);
  float* exb  = (float*)p;            p += (size_t)(E + Nn) * 8 * sizeof(float);
  _Float16* xh = (_Float16*)p;        p += N128 * sizeof(_Float16); // A operand (f16)
  _Float16* wp = (_Float16*)p;        p += 16384 * sizeof(_Float16); // packed W

  const dim3 blk(256);
  const unsigned gN128 = (unsigned)((N128 + 255) / 256);
  const unsigned gNH8  = (unsigned)((NH + 255) / 256);
  const long long eh = (long long)(E + Nn) * 8;
  const long long es = (long long)(E + Nn) * 32;
  const unsigned gEH  = (unsigned)((eh + 255) / 256);
  const unsigned gES  = (unsigned)((es + 255) / 256);
  const unsigned gGemm = (unsigned)((Nn + 127) / 128);
  const unsigned gCvt  = (unsigned)((N128 / 4 + 255) / 256);

  auto edge_phase = [&](const float* as, const float* ad) {
    gat_coeffs<<<gNH8, blk, 0, stream>>>(hbuf, as, ad, asrc, adst, Nn);
    gat_init<<<gN128, blk, 0, stream>>>(mbuf, den, accb, (int)NH, (int)N128);
    gat_edge_max<<<gEH, blk, 0, stream>>>(ei, E, Nn, asrc, adst, exb, mbuf);
    gat_edge_exp<<<gEH, blk, 0, stream>>>(ei, E, Nn, mbuf, exb, den);
    gat_scatter<<<gES, blk, 0, stream>>>(ei, E, Nn, exb, den, hbuf, accb);
  };

  // ---- layer 1 ----
  gat_cvt_f16<<<gCvt, blk, 0, stream>>>(x, xh, N128 / 4);
  gat_pack_w<<<64, blk, 0, stream>>>(W1, wp);
  gat_gemm_wmma<<<gGemm, blk, 0, stream>>>(xh, wp, hbuf, Nn);
  edge_phase(as1, ad1);
  gat_finalize_f16<<<gN128, blk, 0, stream>>>(accb, b1, xh, Nn); // xh := relu(out1+b1)

  // ---- layer 2 ----
  gat_pack_w<<<64, blk, 0, stream>>>(W2, wp);
  gat_gemm_wmma<<<gGemm, blk, 0, stream>>>(xh, wp, hbuf, Nn);
  edge_phase(as2, ad2);
  gat_finalize_f32<<<gN128, blk, 0, stream>>>(accb, b2, (float*)d_out, Nn);
}
